// LSTMWithAdditiveAttention_84628035601007
// MI455X (gfx1250) — matrix operand order repaired and checked by reading, not compiled, not run on hardware
//
#include <hip/hip_runtime.h>

// Problem constants (from reference)
#define BB  128   // batch
#define NN  196   // num feature vectors
#define TT  32    // time steps
#define ED  1024  // embed dim
#define FDV 1024  // feature dim
#define HD  1024  // hidden
#define AD  1024  // attention dim
#define GD  4096  // 4*H gates
#define XD  2048  // E + FD

typedef __attribute__((ext_vector_type(16))) __bf16 v16bf;
typedef __attribute__((ext_vector_type(8)))  __bf16 v8bf;
typedef __attribute__((ext_vector_type(8)))  float  v8f;
typedef int v4i_ __attribute__((vector_size(16)));   // <4 x i32> for async-LDS builtin

__device__ __forceinline__ float sigmoid_f(float x) { return 1.f / (1.f + __expf(-x)); }
__device__ __forceinline__ float tanh_f(float x)    { return 2.f / (1.f + __expf(-2.f * x)) - 1.f; }

// ---------------------------------------------------------------------------
// CDNA5 async global->LDS copy (ASYNCcnt-tracked), builtin if available,
// inline asm (exact gfx1250 mnemonic) otherwise. Device-pass only.
// Builtin signature (from hipcc diagnostic): (v4i* global_src, v4i* lds_dst,
// imm offset, imm cpol).
// ---------------------------------------------------------------------------
#if defined(__HIP_DEVICE_COMPILE__)
#if __has_builtin(__builtin_amdgcn_global_load_async_to_lds_b128)
#define HAVE_ASYNC_BUILTIN 1
#endif
#if __has_builtin(__builtin_amdgcn_s_wait_asynccnt)
#define HAVE_WAITA_BUILTIN 1
#endif
#endif

__device__ __forceinline__ void async_copy16(void* lptr, const void* gptr) {
#if defined(__HIP_DEVICE_COMPILE__)
#if defined(HAVE_ASYNC_BUILTIN)
  __builtin_amdgcn_global_load_async_to_lds_b128((v4i_*)gptr, (v4i_*)lptr, 0, 0);
#else
  asm volatile("global_load_async_to_lds_b128 %0, %1, off"
               :: "v"((unsigned)(unsigned long long)lptr),
                  "v"((unsigned long long)gptr)
               : "memory");
#endif
#else
  (void)lptr; (void)gptr;
#endif
}

template <int N_>
__device__ __forceinline__ void wait_async() {
#if defined(__HIP_DEVICE_COMPILE__)
#if defined(HAVE_WAITA_BUILTIN)
  __builtin_amdgcn_s_wait_asynccnt((unsigned short)N_);
#else
  asm volatile("s_wait_asynccnt %0" :: "n"(N_) : "memory");
#endif
#endif
}

// ---------------------------------------------------------------------------
// WMMA GEMM:  C[M,N] = A0[M,K0] * Bt0[N,K0]^T (+ A1[M,K1]*Bt1[N,K1]^T) (+bias)
// Block = 8 waves. Each wave owns one 16-wide N-tile and marches down an
// M-strip of MT tiles, so its B fragment (32B contiguous/lane) is reused MT
// times. The A strip (MT*16 rows x 32 k) is staged in LDS (async, double
// buffered) and shared by all 8 waves. Bt stored [N,K] row-major (bf16).
// grid: x = N/128, y = M/(MT*16)
// ---------------------------------------------------------------------------
template <bool A0F32, bool OUTBF16, int MT>
__global__ __launch_bounds__(256) void gemm_wmma_kernel(
    const void* __restrict__ A0v, int K0, const __bf16* __restrict__ Bt0,
    const __bf16* __restrict__ A1, int K1, const __bf16* __restrict__ Bt1,
    const float* __restrict__ bias, void* __restrict__ Cv, int N)
{
  constexpr int ROWS   = MT * 16;
  constexpr int CHUNKS = ROWS * 4;        // 16-byte chunks per 32-k slab
  constexpr int PT     = CHUNKS / 256;    // async ops issued per thread per slab

  __shared__ __bf16 As[2][ROWS * 32];

  const int tid  = threadIdx.x;
  const int lane = tid & 31;
  const int wave = tid >> 5;
  const int m0   = blockIdx.y * ROWS;
  const int n0   = blockIdx.x * 128 + wave * 16;
  const int col  = lane & 15;             // B column / C column / A row-in-tile
  const int kg   = lane >> 4;             // lane half selects K sub-block

  v8f acc[MT] = {};

  // ---- pair 0: f32 A (synchronous staged convert) or bf16 A (async) ----
  if constexpr (A0F32) {
    const float* A = (const float*)A0v;
    const __bf16* Bp = Bt0 + (size_t)(n0 + col) * K0 + kg * 16;
    for (int k0 = 0; k0 < K0; k0 += 32) {
      __syncthreads();                    // readers of previous slab done
#pragma unroll
      for (int c = tid; c < CHUNKS; c += 256) {
        const int rw = c >> 2, seg = (c & 3) * 8;
        const float* g = A + (size_t)(m0 + rw) * K0 + k0 + seg;
        __bf16* l = &As[0][rw * 32 + seg];
#pragma unroll
        for (int e = 0; e < 8; ++e) l[e] = (__bf16)g[e];
      }
      __syncthreads();
      const v16bf bfr = *(const v16bf*)(Bp + k0);
#pragma unroll
      for (int mt = 0; mt < MT; ++mt) {
        const __bf16* ap = &As[0][(mt * 16 + col) * 32 + kg * 16];
        v8bf lo = *(const v8bf*)ap;
        v8bf hi = *(const v8bf*)(ap + 8);
        v16bf a;
#pragma unroll
        for (int i = 0; i < 8; ++i) { a[i] = lo[i]; a[i + 8] = hi[i]; }
        acc[mt] = __builtin_amdgcn_wmma_f32_16x16x32_bf16(false, a, false, bfr,
                                                          (short)0, acc[mt], false, false);
      }
    }
    __syncthreads();
  }

  auto run_pair_bf16 = [&](const __bf16* A, int K, const __bf16* Bt) {
    const __bf16* Bp = Bt + (size_t)(n0 + col) * K + kg * 16;
    auto stage = [&](int bf, int k0) {
#pragma unroll
      for (int c = tid; c < CHUNKS; c += 256) {
        const int rw = c >> 2, within = (c & 3) * 16;   // bytes within 64B row
        const char* g = (const char*)(A + (size_t)(m0 + rw) * K + k0) + within;
        char* l = (char*)&As[bf][0] + rw * 64 + within;
        async_copy16(l, g);
      }
    };
    const int KT = K / 32;
    stage(0, 0);
    for (int kk = 0; kk < KT; ++kk) {
      const int bf = kk & 1;
      if (kk + 1 < KT) { stage(bf ^ 1, (kk + 1) * 32); wait_async<PT>(); }
      else             { wait_async<0>(); }
      __syncthreads();                    // slab kk resident for all waves
      const v16bf bfr = *(const v16bf*)(Bp + kk * 32);
#pragma unroll
      for (int mt = 0; mt < MT; ++mt) {
        const __bf16* ap = &As[bf][(mt * 16 + col) * 32 + kg * 16];
        v8bf lo = *(const v8bf*)ap;
        v8bf hi = *(const v8bf*)(ap + 8);
        v16bf a;
#pragma unroll
        for (int i = 0; i < 8; ++i) { a[i] = lo[i]; a[i + 8] = hi[i]; }
        acc[mt] = __builtin_amdgcn_wmma_f32_16x16x32_bf16(false, a, false, bfr,
                                                          (short)0, acc[mt], false, false);
      }
      __syncthreads();                    // readers done before buffer reuse
    }
  };

  if constexpr (!A0F32) run_pair_bf16((const __bf16*)A0v, K0, Bt0);
  if (A1 != nullptr)    run_pair_bf16(A1, K1, Bt1);

  // ---- epilogue: VGPR r of tile mt holds M = m0+mt*16 + r + 8*kg, N = n0+col
  const float bv = bias ? bias[n0 + col] : 0.f;
#pragma unroll
  for (int mt = 0; mt < MT; ++mt) {
#pragma unroll
    for (int r = 0; r < 8; ++r) {
      const size_t off = (size_t)(m0 + mt * 16 + kg * 8 + r) * N + n0 + col;
      const float val = acc[mt][r] + bv;
      if (OUTBF16) ((__bf16*)Cv)[off] = (__bf16)val;
      else         ((float*)Cv)[off]  = val;
    }
  }
}

// ---------------------------------------------------------------------------
// Weight prep kernels
// ---------------------------------------------------------------------------
__global__ void transpose_bf16_kernel(const float* __restrict__ src, __bf16* __restrict__ dst,
                                      int R, int C) {  // src[R,C] -> dst[C,R] (bf16)
  __shared__ float tile[32][33];
  const int c0 = blockIdx.x * 32, r0 = blockIdx.y * 32;
  const int tx = threadIdx.x & 31, ty = threadIdx.x >> 5;  // 32x8
  for (int i = ty; i < 32; i += 8) tile[i][tx] = src[(size_t)(r0 + i) * C + c0 + tx];
  __syncthreads();
  for (int i = ty; i < 32; i += 8) dst[(size_t)(c0 + i) * R + r0 + tx] = (__bf16)tile[tx][i];
}

__global__ void convert_bf16_kernel(const float* __restrict__ src, __bf16* __restrict__ dst, int n) {
  int i = blockIdx.x * 256 + threadIdx.x;
  if (i < n) dst[i] = (__bf16)src[i];
}

__global__ void addconvert_bf16_kernel(const float* __restrict__ a, const float* __restrict__ b,
                                       __bf16* __restrict__ dst, int n) {
  int i = blockIdx.x * 256 + threadIdx.x;
  if (i < n) dst[i] = (__bf16)(a[i] + b[i]);
}

__global__ void biassum_kernel(const float* bi0, const float* bh0, const float* bi1,
                               const float* bh1, float* b0s, float* b1s) {
  int i = blockIdx.x * 256 + threadIdx.x;
  if (i < GD) { b0s[i] = bi0[i] + bh0[i]; b1s[i] = bi1[i] + bh1[i]; }
}

__global__ void init_state_kernel(__bf16* hB, float* c0) {
  int i = blockIdx.x * 256 + threadIdx.x;  // B*H
  hB[i] = (__bf16)0.f;
  c0[i] = 0.f;
}

// ---------------------------------------------------------------------------
// Attention: scores[b,n] = sum_a tanh(f_proj[b,n,a] + hWh[b,a]) * v[a]
// f_proj is bf16 and L2-resident (51 MB << 192 MB L2).
// ---------------------------------------------------------------------------
__global__ __launch_bounds__(128) void scores_kernel(const __bf16* __restrict__ fproj,
                                                     const float* __restrict__ hWh,
                                                     const float* __restrict__ v,
                                                     float* __restrict__ scores) {
  const int row = blockIdx.x;            // 0 .. B*N-1
  const int b   = row / NN;
  const __bf16* fp = fproj + (size_t)row * AD;
  const float*  hw = hWh + (size_t)b * AD;
  float acc = 0.f;
  for (int a = threadIdx.x; a < AD; a += 128)
    acc += tanh_f((float)fp[a] + hw[a]) * v[a];
  __shared__ float red[128];
  red[threadIdx.x] = acc;
  __syncthreads();
  for (int s = 64; s > 0; s >>= 1) {
    if (threadIdx.x < s) red[threadIdx.x] += red[threadIdx.x + s];
    __syncthreads();
  }
  if (threadIdx.x == 0) scores[row] = red[0];
}

// softmax over N per batch (recomputed per f-chunk, N=196 is tiny), then
// attn[b,f] = sum_n alpha[n] * feats[b,n,f]; writes x = concat(embed_t, attn)
__global__ __launch_bounds__(256) void attn_x_kernel(const float* __restrict__ scores,
                                                     const float* __restrict__ feats,
                                                     const float* __restrict__ emb,
                                                     __bf16* __restrict__ xB, int t) {
  __shared__ float red[256];
  __shared__ float alpha[NN];
  const int b   = blockIdx.y;
  const int tid = threadIdx.x;
  float s = (tid < NN) ? scores[b * NN + tid] : -3.0e38f;
  red[tid] = s;
  __syncthreads();
  for (int st = 128; st > 0; st >>= 1) {
    if (tid < st) red[tid] = fmaxf(red[tid], red[tid + st]);
    __syncthreads();
  }
  const float mx = red[0];
  __syncthreads();
  float e = (tid < NN) ? __expf(s - mx) : 0.f;
  red[tid] = e;
  __syncthreads();
  for (int st = 128; st > 0; st >>= 1) {
    if (tid < st) red[tid] += red[tid + st];
    __syncthreads();
  }
  const float inv = 1.f / red[0];
  if (tid < NN) alpha[tid] = e * inv;
  __syncthreads();

  const int f = blockIdx.x * 256 + tid;
  const float* fb = feats + (size_t)b * NN * FDV + f;
  float acc = 0.f;
#pragma unroll 4
  for (int n = 0; n < NN; ++n) acc += alpha[n] * fb[(size_t)n * FDV];
  xB[(size_t)b * XD + ED + f] = (__bf16)acc;
  xB[(size_t)b * XD + f]      = (__bf16)emb[((size_t)b * TT + t) * ED + f];
}

// ---------------------------------------------------------------------------
// LSTM cells (PyTorch gate order i,f,g,o)
// ---------------------------------------------------------------------------
__global__ __launch_bounds__(256) void cell0_kernel(const float* __restrict__ gates,
                                                    float* __restrict__ c, __bf16* __restrict__ hB) {
  const int idx = blockIdx.x * 256 + threadIdx.x;   // B*H
  const int b = idx >> 10, h = idx & 1023;
  const float* g = gates + (size_t)b * GD;
  const float ig = sigmoid_f(g[h]);
  const float fg = sigmoid_f(g[HD + h]);
  const float gg = tanh_f(g[2 * HD + h]);
  const float og = sigmoid_f(g[3 * HD + h]);
  const float cn = fg * c[idx] + ig * gg;
  c[idx]  = cn;                          // c0n: carried state AND layer-1's c input
  hB[idx] = (__bf16)(og * tanh_f(cn));   // h0n (bf16 for next GEMMs)
}

__global__ __launch_bounds__(256) void cell1_kernel(const float* __restrict__ gates,
                                                    const float* __restrict__ c,
                                                    float* __restrict__ out, int t) {
  const int idx = blockIdx.x * 256 + threadIdx.x;
  const int b = idx >> 10, h = idx & 1023;
  const float* g = gates + (size_t)b * GD;
  const float ig = sigmoid_f(g[h]);
  const float fg = sigmoid_f(g[HD + h]);
  const float gg = tanh_f(g[2 * HD + h]);
  const float og = sigmoid_f(g[3 * HD + h]);
  const float cn = fg * c[idx] + ig * gg;     // c1 discarded (not carried)
  out[((size_t)b * TT + t) * HD + h] = og * tanh_f(cn);
}

// ---------------------------------------------------------------------------
extern "C" void kernel_launch(void* const* d_in, const int* in_sizes, int n_in,
                              void* d_out, int out_size, void* d_ws, size_t ws_size,
                              hipStream_t stream) {
  (void)in_sizes; (void)n_in; (void)out_size; (void)ws_size;
  const float* feats = (const float*)d_in[0];
  const float* emb   = (const float*)d_in[1];
  const float* Wf    = (const float*)d_in[2];
  const float* Wh    = (const float*)d_in[3];
  const float* b_att = (const float*)d_in[4];
  const float* vvec  = (const float*)d_in[5];
  const float* Wih0  = (const float*)d_in[6];
  const float* Whh0  = (const float*)d_in[7];
  const float* bih0  = (const float*)d_in[8];
  const float* bhh0  = (const float*)d_in[9];
  const float* Wih1  = (const float*)d_in[10];
  const float* Whh1  = (const float*)d_in[11];
  const float* bih1  = (const float*)d_in[12];
  const float* bhh1  = (const float*)d_in[13];
  float* out = (float*)d_out;

  // workspace carve-up (~95 MB total)
  char* p = (char*)d_ws;
  auto alloc = [&](size_t bytes) -> char* {
    char* r = p; p += (bytes + 255) & ~(size_t)255; return r;
  };
  __bf16* fprojB = (__bf16*)alloc((size_t)BB * NN * AD * 2);  // 51.4 MB, L2-resident
  __bf16* WfT    = (__bf16*)alloc((size_t)AD * FDV * 2);
  __bf16* WhT    = (__bf16*)alloc((size_t)AD * HD * 2);
  __bf16* Wih0B  = (__bf16*)alloc((size_t)GD * XD * 2);
  __bf16* Whh0B  = (__bf16*)alloc((size_t)GD * HD * 2);
  __bf16* W1sumB = (__bf16*)alloc((size_t)GD * HD * 2);
  float*  b0sum  = (float*)alloc(GD * 4);
  float*  b1sum  = (float*)alloc(GD * 4);
  float*  hWh    = (float*)alloc((size_t)BB * AD * 4);
  float*  scores = (float*)alloc((size_t)BB * NN * 4);
  __bf16* xB     = (__bf16*)alloc((size_t)BB * XD * 2);
  __bf16* hB     = (__bf16*)alloc((size_t)BB * HD * 2);
  float*  c0     = (float*)alloc((size_t)BB * HD * 4);
  float*  gates0 = (float*)alloc((size_t)BB * GD * 4);
  float*  gates1 = (float*)alloc((size_t)BB * GD * 4);

  // --- one-time weight prep ---
  transpose_bf16_kernel<<<dim3(AD / 32, FDV / 32), 256, 0, stream>>>(Wf, WfT, FDV, AD);
  transpose_bf16_kernel<<<dim3(AD / 32, HD / 32), 256, 0, stream>>>(Wh, WhT, HD, AD);
  convert_bf16_kernel<<<(GD * XD) / 256, 256, 0, stream>>>(Wih0, Wih0B, GD * XD);
  convert_bf16_kernel<<<(GD * HD) / 256, 256, 0, stream>>>(Whh0, Whh0B, GD * HD);
  addconvert_bf16_kernel<<<(GD * HD) / 256, 256, 0, stream>>>(Wih1, Whh1, W1sumB, GD * HD);
  biassum_kernel<<<GD / 256, 256, 0, stream>>>(bih0, bhh0, bih1, bhh1, b0sum, b1sum);
  init_state_kernel<<<(BB * HD) / 256, 256, 0, stream>>>(hB, c0);

  // --- f_proj = feats @ Wf + b_att -> bf16 [B*N, A]; MT=8 (128-row strips) ---
  gemm_wmma_kernel<true, true, 8><<<dim3(AD / 128, (BB * NN) / 128), 256, 0, stream>>>(
      (const void*)feats, FDV, WfT, nullptr, 0, nullptr, b_att, (void*)fprojB, AD);

  // --- sequential recurrence; MT=4 (64-row strips, 2x parallelism at M=128) ---
  for (int t = 0; t < TT; ++t) {
    // hWh = h0 @ Wh   [128,1024]x[1024,1024]
    gemm_wmma_kernel<false, false, 4><<<dim3(AD / 128, BB / 64), 256, 0, stream>>>(
        (const void*)hB, HD, WhT, nullptr, 0, nullptr, nullptr, (void*)hWh, AD);
    // additive-attention energies + reduce over A
    scores_kernel<<<BB * NN, 128, 0, stream>>>(fprojB, hWh, vvec, scores);
    // softmax + context vector + build x = [embed_t, attn] in bf16
    attn_x_kernel<<<dim3(FDV / 256, BB), 256, 0, stream>>>(scores, feats, emb, xB, t);
    // gates0 = x @ Wih0^T + h0 @ Whh0^T + b  (dual-K WMMA GEMM, N=4096)
    gemm_wmma_kernel<false, false, 4><<<dim3(GD / 128, BB / 64), 256, 0, stream>>>(
        (const void*)xB, XD, Wih0B, hB, HD, Whh0B, b0sum, (void*)gates0, GD);
    cell0_kernel<<<(BB * HD) / 256, 256, 0, stream>>>(gates0, c0, hB);
    // gates1 = h0n @ (Wih1 + Whh1)^T + b   (weights pre-summed: one GEMM)
    gemm_wmma_kernel<false, false, 4><<<dim3(GD / 128, BB / 64), 256, 0, stream>>>(
        (const void*)hB, HD, W1sumB, nullptr, 0, nullptr, b1sum, (void*)gates1, GD);
    cell1_kernel<<<(BB * HD) / 256, 256, 0, stream>>>(gates1, c0, out, t);
  }
}
